// SAFFSlotAttention_9010841387367
// MI455X (gfx1250) — compile-verified, matrix operand order checked by reading, not run on hardware
//
#include <hip/hip_runtime.h>

#define NSLOT 5
#define NITER 5

typedef __attribute__((ext_vector_type(16))) __bf16 v16bf;
typedef __attribute__((ext_vector_type(8)))  __bf16 v8bf;
typedef __attribute__((ext_vector_type(8)))  float  v8f;

__device__ inline float sigmf(float x) { return 1.0f / (1.0f + __expf(-x)); }

// Load one 16x32 bf16 WMMA fragment (A-style / mirrored-B-style layout).
// rowbase already points at this lane's row; lane selects the K half.
__device__ inline v16bf frag_ld(const __bf16* rowbase, int k0, int lane) {
  const __bf16* p = rowbase + k0 + ((lane & 16) ? 8 : 0);
  v8bf lo = *(const v8bf*)p;
  v8bf hi = *(const v8bf*)(p + 16);
  v16bf r;
#pragma unroll
  for (int i = 0; i < 8; ++i) { r[i] = lo[i]; r[i + 8] = hi[i]; }
  return r;
}

__device__ inline v8f wmma_bf16(v16bf a, v16bf b, v8f c) {
  return __builtin_amdgcn_wmma_f32_16x16x32_bf16(false, a, false, b, (short)0, c,
                                                 false, false);
}

// ---- CDNA5 async global->LDS DMA (ASYNCcnt path), inline asm (portable
// across ROCm7.2 / amdgpu-toolchain builtin-arity differences).
__device__ inline unsigned lds_off_u32(const void* p) {
  // generic LDS pointer low 32 bits == LDS byte offset (aperture mapping)
  return (unsigned)(unsigned long long)p;
}
__device__ inline void async_g2l_b128(unsigned lds_off, const void* gptr) {
  asm volatile("global_load_async_to_lds_b128 %0, %1, off"
               :: "v"(lds_off), "v"((unsigned long long)gptr)
               : "memory");
}
template <int N>
__device__ inline void wait_async() {
  asm volatile("s_wait_asynccnt %0" :: "i"(N) : "memory");
}

// ---------------------------------------------------------------- K0: weights -> bf16
__global__ void __launch_bounds__(256) k0_cvt(const float* __restrict__ Wk,
                                              const float* __restrict__ Wv,
                                              __bf16* __restrict__ Wkv) {
  int id = blockIdx.x * 256 + threadIdx.x;  // 0..131071
  int row = id >> 8, c = id & 255;
  float v = (row < 256) ? Wk[row * 256 + c] : Wv[(row - 256) * 256 + c];
  Wkv[id] = (__bf16)v;
}

// ---------------------------------------------------------------- init slots
__global__ void __launch_bounds__(256) k_init(const float* __restrict__ noise,
                                              const float* __restrict__ mu,
                                              const float* __restrict__ sigma,
                                              float* __restrict__ slots) {
  int id = blockIdx.x * 256 + threadIdx.x;  // 0..81919
  int c = id & 255;
  slots[id] = mu[c] + sigma[c] * noise[id];
}

// ---------------------------------------------------------------- K1: LN stats (wave per row)
__global__ void __launch_bounds__(256) k1_stats(const float* __restrict__ patches,
                                                float* __restrict__ stats) {
  int tid = threadIdx.x;
  size_t row = (size_t)blockIdx.x * 8 + (tid >> 5);
  int lane = tid & 31;
  const float4* p = (const float4*)(patches + row * 256 + lane * 8);
  float4 a = p[0], c = p[1];
  float s = a.x + a.y + a.z + a.w + c.x + c.y + c.z + c.w;
  float q = a.x * a.x + a.y * a.y + a.z * a.z + a.w * a.w +
            c.x * c.x + c.y * c.y + c.z * c.z + c.w * c.w;
#pragma unroll
  for (int off = 16; off > 0; off >>= 1) {
    s += __shfl_xor(s, off);
    q += __shfl_xor(q, off);
  }
  if (lane == 0) {
    float mean = s * (1.0f / 256.0f);
    float var = q * (1.0f / 256.0f) - mean * mean;
    stats[row * 2 + 0] = mean;
    stats[row * 2 + 1] = rsqrtf(var + 1e-5f);
  }
}

// ---------------------------------------------------------------- K2: fused LN + [k|v] GEMM
// C(M=262144, N=512) = LN(patches) @ Wkv^T, K=256. Block: M-tile 128, 8 waves.
// n<256 -> kmat row-major (n-major); n>=256 -> v stored transposed (c, patch).
__global__ void __launch_bounds__(256) k2_kvgemm(
    const float* __restrict__ patches, const float* __restrict__ stats,
    const float* __restrict__ g, const float* __restrict__ bvec,
    const __bf16* __restrict__ Wkv, __bf16* __restrict__ kmat,
    __bf16* __restrict__ vt) {
  __shared__ __bf16 As[128 * 256];  // 64 KB LDS tile (LN applied, bf16)
  const int tid = threadIdx.x;
  const size_t m0 = (size_t)blockIdx.x * 128;
  const int b = (int)(m0 >> 12);

#pragma unroll 4
  for (int j = 0; j < 32; ++j) {
    int vi = j * 256 + tid;  // float4 index, 8192 total
    int row = vi >> 6;
    int c4 = vi & 63;
    int c = c4 * 4;
    const float4 x = ((const float4*)(patches + (m0 + row) * 256))[c4];
    float mean = stats[(m0 + row) * 2 + 0];
    float rstd = stats[(m0 + row) * 2 + 1];
    __bf16* dst = As + row * 256 + c;
    dst[0] = (__bf16)((x.x - mean) * rstd * g[c + 0] + bvec[c + 0]);
    dst[1] = (__bf16)((x.y - mean) * rstd * g[c + 1] + bvec[c + 1]);
    dst[2] = (__bf16)((x.z - mean) * rstd * g[c + 2] + bvec[c + 2]);
    dst[3] = (__bf16)((x.w - mean) * rstd * g[c + 3] + bvec[c + 3]);
  }
  __syncthreads();

  const int lane = tid & 31;
  const int w = tid >> 5;
  const int l = lane & 15;
  const bool hi = (lane & 16) != 0;
  const __bf16* arow = As + (w * 16 + l) * 256;

  // Hoist the wave's whole A row-strip (16x256) into registers: one LDS read,
  // reused across all 8 N-chunks; frees the k-loop to be pure global-load+WMMA.
  v16bf afr[8];
#pragma unroll
  for (int ks = 0; ks < 8; ++ks) afr[ks] = frag_ld(arow, ks * 32, lane);

  for (int nc = 0; nc < 8; ++nc) {
    const int c0 = nc * 64;
    v8f acc[4] = {};
#pragma unroll
    for (int ks = 0; ks < 8; ++ks) {
      const int k0 = ks * 32;
#pragma unroll
      for (int t = 0; t < 4; ++t) {
        v16bf bf = frag_ld(Wkv + (size_t)(c0 + 16 * t + l) * 256, k0, lane);
        acc[t] = wmma_bf16(afr[ks], bf, acc[t]);
      }
    }
#pragma unroll
    for (int t = 0; t < 4; ++t) {
      int n = c0 + 16 * t + l;
#pragma unroll
      for (int r = 0; r < 8; ++r) {
        size_t m = m0 + (size_t)w * 16 + r + (hi ? 8 : 0);
        __bf16 val = (__bf16)acc[t][r];
        if (n < 256) {
          kmat[m * 256 + n] = val;
        } else {
          size_t nb = m - (size_t)b * 4096;
          vt[((size_t)b * 256 + (n - 256)) * 4096 + nb] = val;
        }
      }
    }
  }
}

// ---------------------------------------------------------------- K3: slot LN + q projection
__global__ void __launch_bounds__(256) k3_q(
    const float* __restrict__ slots, const float* __restrict__ Wq,
    const float* __restrict__ g, const float* __restrict__ bvec,
    __bf16* __restrict__ qbf, float* __restrict__ rowsum) {
  __shared__ float sl[NSLOT][256];
  __shared__ float red1[256], red2[256];
  const int b = blockIdx.x, d = threadIdx.x;
  if (d < 8) rowsum[b * 8 + d] = 0.0f;  // zeroed for this iteration's K5
  for (int l = 0; l < NSLOT; ++l) {
    float v = slots[((size_t)b * NSLOT + l) * 256 + d];
    red1[d] = v;
    red2[d] = v * v;
    __syncthreads();
    for (int s2 = 128; s2 > 0; s2 >>= 1) {
      if (d < s2) { red1[d] += red1[d + s2]; red2[d] += red2[d + s2]; }
      __syncthreads();
    }
    float mean = red1[0] * (1.0f / 256.0f);
    float var = red2[0] * (1.0f / 256.0f) - mean * mean;
    float rstd = rsqrtf(var + 1e-5f);
    sl[l][d] = (v - mean) * rstd * g[d] + bvec[d];
    __syncthreads();
  }
  float acc[NSLOT] = {};
  const float* wr = Wq + (size_t)d * 256;
  for (int j = 0; j < 256; ++j) {
    float wv = wr[j];
#pragma unroll
    for (int l = 0; l < NSLOT; ++l) acc[l] += sl[l][j] * wv;
  }
  __bf16* qo = qbf + (size_t)b * 16 * 256 + d;
#pragma unroll
  for (int l = 0; l < NSLOT; ++l) qo[l * 256] = (__bf16)acc[l];
#pragma unroll
  for (int l = NSLOT; l < 16; ++l) qo[l * 256] = (__bf16)0.0f;  // zero-pad for WMMA
}

// ---------------------------------------------------------------- K4: dots = q @ k^T * SCALE
__global__ void __launch_bounds__(256) k4_dots(const __bf16* __restrict__ qbf,
                                               const __bf16* __restrict__ kmat,
                                               float* __restrict__ dots) {
  __shared__ __bf16 qs[16 * 256];  // 8 KB: q tile for this batch
  const int tid = threadIdx.x;
  const int b = blockIdx.x >> 5;
  const int nblk = blockIdx.x & 31;

  // async DMA the q tile into LDS (512 x 16B chunks, 2 per thread)
  const unsigned qs_base = lds_off_u32(qs);
  const char* qsrc = (const char*)(qbf + (size_t)b * 16 * 256);
#pragma unroll
  for (int j = 0; j < 2; ++j) {
    int chunk = j * 256 + tid;  // 0..511
    async_g2l_b128(qs_base + (unsigned)chunk * 16, qsrc + (size_t)chunk * 16);
  }
  wait_async<0>();
  __syncthreads();

  const int lane = tid & 31;
  const int w = tid >> 5;
  const int l = lane & 15;
  const bool hi = (lane & 16) != 0;
  const int n0 = nblk * 128 + w * 16;

  v8f acc = {};
  const __bf16* krow = kmat + ((size_t)b * 4096 + n0 + l) * 256;
#pragma unroll
  for (int ks = 0; ks < 8; ++ks) {
    v16bf a = frag_ld(qs + l * 256, ks * 32, lane);
    v16bf bf = frag_ld(krow, ks * 32, lane);
    acc = wmma_bf16(a, bf, acc);
  }
  if (!hi) {
#pragma unroll
    for (int r = 0; r < NSLOT; ++r)
      dots[((size_t)b * NSLOT + r) * 4096 + n0 + l] = acc[r] * 0.0625f;  // DIM^-0.5
  }
}

// ---------------------------------------------------------------- K5: softmax over slots + rowsum
__global__ void __launch_bounds__(256) k5_softmax(const float* __restrict__ dots,
                                                  __bf16* __restrict__ attnbf,
                                                  float* __restrict__ rowsum,
                                                  float* __restrict__ attn_out) {
  __shared__ float ssum[NSLOT];
  int tid = threadIdx.x;
  size_t id = (size_t)blockIdx.x * 256 + tid;
  int b = (int)(id >> 12), n = (int)(id & 4095);
  if (tid < NSLOT) ssum[tid] = 0.0f;
  __syncthreads();
  const float* dp = dots + (size_t)b * NSLOT * 4096 + n;
  float v[NSLOT];
#pragma unroll
  for (int s = 0; s < NSLOT; ++s) v[s] = dp[(size_t)s * 4096];
  float m = v[0];
#pragma unroll
  for (int s = 1; s < NSLOT; ++s) m = fmaxf(m, v[s]);
  float sum = 0.0f;
#pragma unroll
  for (int s = 0; s < NSLOT; ++s) { v[s] = __expf(v[s] - m); sum += v[s]; }
  float inv = 1.0f / sum;
#pragma unroll
  for (int s = 0; s < NSLOT; ++s) {
    float a = v[s] * inv;
    attnbf[((size_t)b * 16 + s) * 4096 + n] = (__bf16)a;
    if (attn_out) attn_out[((size_t)b * NSLOT + s) * 4096 + n] = a;
    atomicAdd(&ssum[s], a);
  }
  __syncthreads();
  if (tid < NSLOT) atomicAdd(&rowsum[b * 8 + tid], ssum[tid]);
}

// ---------------------------------------------------------------- K6: updates = (attn/rowsum) @ v
// A (attn, 16x4096) streamed global->LDS with double-buffered async DMA while
// WMMA consumes the previous 32KB stage; B (v^T) streamed straight from HBM.
__global__ void __launch_bounds__(256) k6_updates(const __bf16* __restrict__ attnbf,
                                                  const __bf16* __restrict__ vt,
                                                  const float* __restrict__ rowsum,
                                                  float* __restrict__ updates) {
  __shared__ __bf16 abuf[2][16 * 1024];  // 2 x 32 KB double buffer
  const int b = blockIdx.x;
  const int tid = threadIdx.x;
  const int lane = tid & 31;
  const int w = tid >> 5;
  const int l = lane & 15;
  const bool hi = (lane & 16) != 0;

  const __bf16* asrc = attnbf + (size_t)b * 16 * 4096;
  const unsigned abase0 = lds_off_u32(abuf[0]);
  const unsigned abase1 = lds_off_u32(abuf[1]);

  auto issue_stage = [&](int kc) {
    const unsigned base = (kc & 1) ? abase1 : abase0;
    // 16 rows x 1024 cols bf16 = 32KB = 2048 x 16B chunks, 8 per thread
#pragma unroll
    for (int j = 0; j < 8; ++j) {
      int chunk = j * 256 + tid;      // 0..2047
      int row = chunk >> 7;           // 128 chunks per row
      int col = (chunk & 127) * 8;    // element column within stage
      async_g2l_b128(base + (unsigned)chunk * 16,
                     asrc + (size_t)row * 4096 + kc * 1024 + col);
    }
  };

  const __bf16* vrow0 = vt + ((size_t)b * 256 + w * 32 + 0 + l) * 4096;
  const __bf16* vrow1 = vt + ((size_t)b * 256 + w * 32 + 16 + l) * 4096;
  v8f acc[2] = {};

  issue_stage(0);
  for (int kc = 0; kc < 4; ++kc) {
    if (kc < 3) {
      issue_stage(kc + 1);  // overlap next DMA stage with this compute stage
      wait_async<8>();      // in-order: stage kc resident, kc+1 still in flight
    } else {
      wait_async<0>();
    }
    __syncthreads();
    const __bf16* arow = ((kc & 1) ? abuf[1] : abuf[0]) + l * 1024;
    const __bf16* v0p = vrow0 + kc * 1024;
    const __bf16* v1p = vrow1 + kc * 1024;
#pragma unroll 4
    for (int ks = 0; ks < 32; ++ks) {
      const int k0 = ks * 32;
      if ((k0 & 255) == 0) {
        __builtin_prefetch(v0p + k0 + 512, 0, 1);
        __builtin_prefetch(v1p + k0 + 512, 0, 1);
      }
      v16bf a = frag_ld(arow, k0, lane);
      v16bf b0 = frag_ld(v0p, k0, lane);
      v16bf b1 = frag_ld(v1p, k0, lane);
      acc[0] = wmma_bf16(a, b0, acc[0]);
      acc[1] = wmma_bf16(a, b1, acc[1]);
    }
    __syncthreads();  // done reading abuf[kc&1] before next issue targets it
  }

  if (!hi) {
#pragma unroll
    for (int r = 0; r < NSLOT; ++r) {
      float sc = 1.0f / (rowsum[b * 8 + r] + 1e-8f);
#pragma unroll
      for (int t = 0; t < 2; ++t) {
        int c = w * 32 + 16 * t + l;
        updates[((size_t)b * NSLOT + r) * 256 + c] = acc[t][r] * sc;
      }
    }
  }
}

// ---------------------------------------------------------------- K7: residual + Mamba block
__global__ void __launch_bounds__(256) k7_mamba(
    float* __restrict__ slots, const float* __restrict__ updates,
    const float* __restrict__ g, const float* __restrict__ bvec,
    const float* __restrict__ Win, const float* __restrict__ conv_w,
    const float* __restrict__ conv_b, const float* __restrict__ Wx,
    const float* __restrict__ Wdt, const float* __restrict__ bdt,
    const float* __restrict__ A_log, const float* __restrict__ Dp,
    const float* __restrict__ Wout, float* __restrict__ out2) {
  __shared__ float s1[NSLOT][256];
  __shared__ float xln[NSLOT][256];
  __shared__ float xcs[NSLOT][256];
  __shared__ float zsh[NSLOT][256];
  __shared__ float ybs[NSLOT][256];
  __shared__ float xdbl[NSLOT][48];
  __shared__ float red1[256], red2[256];
  const int b = blockIdx.x, d = threadIdx.x;

  for (int l = 0; l < NSLOT; ++l) {
    size_t idx = ((size_t)b * NSLOT + l) * 256 + d;
    float v = slots[idx] + updates[idx];
    s1[l][d] = v;
    red1[d] = v;
    red2[d] = v * v;
    __syncthreads();
    for (int s2 = 128; s2 > 0; s2 >>= 1) {
      if (d < s2) { red1[d] += red1[d + s2]; red2[d] += red2[d + s2]; }
      __syncthreads();
    }
    float mean = red1[0] * (1.0f / 256.0f);
    float var = red2[0] * (1.0f / 256.0f) - mean * mean;
    float rstd = rsqrtf(var + 1e-5f);
    xln[l][d] = (v - mean) * rstd * g[d] + bvec[d];
    __syncthreads();
  }

  // in-projection: xz = x @ Win^T  (thread d owns output channels d and d+256)
  float xseq[NSLOT] = {};
  {
    float za[NSLOT] = {};
    const float* w1 = Win + (size_t)d * 256;
    const float* w2 = Win + (size_t)(d + 256) * 256;
    for (int j = 0; j < 256; ++j) {
      float a = w1[j], c = w2[j];
#pragma unroll
      for (int l = 0; l < NSLOT; ++l) {
        xseq[l] += xln[l][j] * a;
        za[l] += xln[l][j] * c;
      }
    }
#pragma unroll
    for (int l = 0; l < NSLOT; ++l) zsh[l][d] = za[l];
  }
  // depthwise causal conv (K=4, left pad 3) + SiLU
  {
    float cw0 = conv_w[d * 4 + 0], cw1 = conv_w[d * 4 + 1];
    float cw2 = conv_w[d * 4 + 2], cw3 = conv_w[d * 4 + 3];
    float cb = conv_b[d];
#pragma unroll
    for (int l = 0; l < NSLOT; ++l) {
      float a = cb + cw3 * xseq[l];
      if (l >= 1) a += cw2 * xseq[l - 1];
      if (l >= 2) a += cw1 * xseq[l - 2];
      if (l >= 3) a += cw0 * xseq[l - 3];
      xcs[l][d] = a * sigmf(a);
    }
  }
  __syncthreads();
  // x_dbl = xc @ Wx^T : 5 rows x 48 outputs, one thread each
  if (d < NSLOT * 48) {
    int l = d / 48, o = d % 48;
    const float* wr = Wx + (size_t)o * 256;
    float a = 0.0f;
    for (int j = 0; j < 256; ++j) a += xcs[l][j] * wr[j];
    xdbl[l][o] = a;
  }
  __syncthreads();
  // dt = softplus(dt_r @ Wdt^T + bdt)
  float dtl[NSLOT];
  {
    float bv = bdt[d];
#pragma unroll
    for (int l = 0; l < NSLOT; ++l) {
      float a = bv;
#pragma unroll
      for (int j = 0; j < 16; ++j) a += xdbl[l][j] * Wdt[d * 16 + j];
      dtl[l] = (a > 20.0f) ? a : log1pf(__expf(a));
    }
  }
  // selective scan, state 16 wide in registers
  {
    float Av[16];
#pragma unroll
    for (int s = 0; s < 16; ++s) Av[s] = -__expf(A_log[d * 16 + s]);
    float dparam = Dp[d];
    float h[16] = {};
#pragma unroll
    for (int l = 0; l < NSLOT; ++l) {
      float dtt = dtl[l], xt = xcs[l][d];
      float y = 0.0f;
#pragma unroll
      for (int s = 0; s < 16; ++s) {
        h[s] = __expf(dtt * Av[s]) * h[s] + dtt * xt * xdbl[l][16 + s];
        y += h[s] * xdbl[l][32 + s];
      }
      y += dparam * xt;
      float zv = zsh[l][d];
      y *= zv * sigmf(zv);
      ybs[l][d] = y;
    }
  }
  __syncthreads();
  // out-projection + residual
  {
    float o[NSLOT] = {};
    const float* wr = Wout + (size_t)d * 256;
    for (int j = 0; j < 256; ++j) {
      float wv = wr[j];
#pragma unroll
      for (int l = 0; l < NSLOT; ++l) o[l] += ybs[l][j] * wv;
    }
#pragma unroll
    for (int l = 0; l < NSLOT; ++l) {
      size_t idx = ((size_t)b * NSLOT + l) * 256 + d;
      float res = s1[l][d] + o[l];
      slots[idx] = res;
      if (out2) out2[idx] = res;
    }
  }
}

// ================================================================ host launch
extern "C" void kernel_launch(void* const* d_in, const int* in_sizes, int n_in,
                              void* d_out, int out_size, void* d_ws, size_t ws_size,
                              hipStream_t stream) {
  (void)in_sizes; (void)n_in; (void)out_size; (void)ws_size;
  const float* patches = (const float*)d_in[0];
  const float* noise   = (const float*)d_in[1];
  const float* mu      = (const float*)d_in[2];
  const float* sigma   = (const float*)d_in[3];
  const float* ln_in_g = (const float*)d_in[4];
  const float* ln_in_b = (const float*)d_in[5];
  const float* ln_sl_g = (const float*)d_in[6];
  const float* ln_sl_b = (const float*)d_in[7];
  const float* ln_mb_g = (const float*)d_in[8];
  const float* ln_mb_b = (const float*)d_in[9];
  const float* Wq      = (const float*)d_in[10];
  const float* Wk      = (const float*)d_in[11];
  const float* Wv      = (const float*)d_in[12];
  const float* Win     = (const float*)d_in[13];
  const float* conv_w  = (const float*)d_in[14];
  const float* conv_b  = (const float*)d_in[15];
  const float* Wx      = (const float*)d_in[16];
  const float* Wdt     = (const float*)d_in[17];
  const float* bdt     = (const float*)d_in[18];
  const float* A_log   = (const float*)d_in[19];
  const float* D_param = (const float*)d_in[20];
  const float* Wout    = (const float*)d_in[21];

  unsigned char* wsb = (unsigned char*)d_ws;
  size_t off = 0;
  auto carve = [&](size_t bytes) -> void* {
    void* p = wsb + off;
    off += (bytes + 255) & ~(size_t)255;
    return p;
  };
  float*  stats   = (float*)carve((size_t)262144 * 2 * 4);          // mean,rstd per row
  __bf16* Wkv     = (__bf16*)carve((size_t)512 * 256 * 2);          // [Wk;Wv] bf16
  __bf16* kmat    = (__bf16*)carve((size_t)262144 * 256 * 2);       // k, (b*4096+n, c)
  __bf16* vt      = (__bf16*)carve((size_t)64 * 256 * 4096 * 2);    // v^T, (b, c, n)
  float*  slots   = (float*)carve((size_t)64 * NSLOT * 256 * 4);
  __bf16* qbf     = (__bf16*)carve((size_t)64 * 16 * 256 * 2);      // q padded to 16 rows
  float*  dots    = (float*)carve((size_t)64 * NSLOT * 4096 * 4);
  __bf16* attnbf  = (__bf16*)carve((size_t)64 * 16 * 4096 * 2);     // attn padded rows
  float*  rowsum  = (float*)carve((size_t)64 * 8 * 4);
  float*  updates = (float*)carve((size_t)64 * NSLOT * 256 * 4);

  float* out_slots = (float*)d_out;                     // (64,5,256)
  float* out_attn  = (float*)d_out + 64 * NSLOT * 256;  // (64,5,4096)

  k0_cvt<<<512, 256, 0, stream>>>(Wk, Wv, Wkv);
  k_init<<<320, 256, 0, stream>>>(noise, mu, sigma, slots);
  k1_stats<<<32768, 256, 0, stream>>>(patches, stats);
  k2_kvgemm<<<2048, 256, 0, stream>>>(patches, stats, ln_in_g, ln_in_b, Wkv, kmat, vt);

  for (int it = 0; it < NITER; ++it) {
    const bool last = (it == NITER - 1);
    k3_q<<<64, 256, 0, stream>>>(slots, Wq, ln_sl_g, ln_sl_b, qbf, rowsum);
    k4_dots<<<2048, 256, 0, stream>>>(qbf, kmat, dots);
    k5_softmax<<<1024, 256, 0, stream>>>(dots, attnbf, rowsum,
                                         last ? out_attn : (float*)nullptr);
    k6_updates<<<64, 256, 0, stream>>>(attnbf, vt, rowsum, updates);
    k7_mamba<<<64, 256, 0, stream>>>(slots, updates, ln_mb_g, ln_mb_b, Win, conv_w,
                                     conv_b, Wx, Wdt, bdt, A_log, D_param, Wout,
                                     last ? out_slots : (float*)nullptr);
  }
}
